// SinkhornDistance_15006615733809
// MI455X (gfx1250) — compile-verified, matrix operand order checked by reading, not run on hardware
//
#include <hip/hip_runtime.h>
#include <hip/hip_bf16.h>

typedef _Float16 v16h __attribute__((ext_vector_type(16)));
typedef float    v8f  __attribute__((ext_vector_type(8)));

#define NS 8192
#define NT 8192
#define D  64
#define EPS      0.1f
#define INV_EPS  10.0f
#define STABF    1e-8f
#define N_ITER   10
#define ROW_CHUNKS 32
#define ROWS_PER_CHUNK (NS / ROW_CHUNKS)
#define FIN_ELEMS 4
#define FIN_BLOCKS ((NS * (size_t)NT) / (256 * FIN_ELEMS))   // 65536

// ---------- phase 0: row squared norms ----------
__global__ void norms_kernel(const float* __restrict__ S, const float* __restrict__ T,
                             float* __restrict__ s2, float* __restrict__ t2) {
    int i = blockIdx.x * blockDim.x + threadIdx.x;
    const float* base; float* out; int r;
    if (i < NS) { base = S; out = s2; r = i; }        // uniform per block (NS % 256 == 0)
    else        { base = T; out = t2; r = i - NS; }
    const float* row = base + (size_t)r * D;
    float acc = 0.f;
#pragma unroll
    for (int k = 0; k < D; ++k) { float x = row[k]; acc += x * x; }
    out[r] = acc;
}

// ---------- phase 1: Gibbs kernel K = exp(-||s_i - t_j|| / eps) via WMMA ----------
// One wave computes one 16x16 tile of S@T^T (K=64 -> 2x v_wmma_f32_16x16x32_f16),
// then applies s2 + t2 - 2*dot, sqrt, exp and stores the tile.
__global__ void gibbs_wmma_kernel(const float* __restrict__ S, const float* __restrict__ T,
                                  const float* __restrict__ s2, const float* __restrict__ t2,
                                  float* __restrict__ Kmat) {
    const int wavesPerBlock = blockDim.x >> 5;
    int wave = blockIdx.x * wavesPerBlock + (threadIdx.x >> 5);
    int lane = threadIdx.x & 31;
    const int tilesN = NT / 16;               // 512
    int tileM = wave / tilesN;
    int tileN = wave - tileM * tilesN;
    int row0 = tileM << 4;
    int col0 = tileN << 4;
    int half = lane >> 4;                     // 0: lanes 0-15, 1: lanes 16-31
    int mr   = lane & 15;                     // A: row index, B: column index

    v8f acc = {0.f, 0.f, 0.f, 0.f, 0.f, 0.f, 0.f, 0.f};
#pragma unroll
    for (int c = 0; c < 2; ++c) {             // two K=32 chunks cover D=64
        const float* arow = S + (size_t)(row0 + mr) * D + c * 32;
        const float* brow = T + (size_t)(col0 + mr) * D + c * 32;
        v16h a, b;
#pragma unroll
        for (int h = 0; h < 16; ++h) {
            // 16-bit A-matrix 16x32 layout: lanes 0-15 hold K {0..7, 16..23},
            // lanes 16-31 hold K {8..15, 24..31}; B symmetric (lane = column).
            int k = (h < 8 ? h : h + 8) + (half ? 8 : 0);
            a[h] = (_Float16)arow[k];
            b[h] = (_Float16)brow[k];
        }
        acc = __builtin_amdgcn_wmma_f32_16x16x32_f16(
            /*neg_a=*/false, a, /*neg_b=*/false, b,
            /*c_mod=*/(short)0, acc, /*reuse_a=*/false, /*reuse_b=*/false);
    }

    // C/D layout: VGPR d -> M = d + 8*half, N = lane & 15
    float tn = t2[col0 + mr];
#pragma unroll
    for (int d = 0; d < 8; ++d) {
        int M = d + (half << 3);
        float sq   = s2[row0 + M] + tn - 2.0f * acc[d];
        float cost = sqrtf(fmaxf(sq, 0.0f));
        float kv   = __expf(-cost * INV_EPS);
        Kmat[(size_t)(row0 + M) * NT + (col0 + mr)] = kv;
    }
}

// ---------- phase 2: Sinkhorn iterations ----------
__global__ void init_uv(float* __restrict__ u, float* __restrict__ v) {
    int i = blockIdx.x * blockDim.x + threadIdx.x;
    if (i < NS) u[i] = 1.0f; else v[i - NS] = 1.0f;
}

// u[i] = (1/NS) / (sum_j K[i,j] * v[j] + stab) ; one wave per row (coalesced b32)
__global__ void row_matvec_u(const float* __restrict__ Kmat, const float* __restrict__ vin,
                             float* __restrict__ uout) {
    int wave = blockIdx.x * (blockDim.x >> 5) + (threadIdx.x >> 5);
    int lane = threadIdx.x & 31;
    const float* row = Kmat + (size_t)wave * NT;
    float acc = 0.f;
#pragma unroll 8
    for (int j = lane; j < NT; j += 32)
        acc += row[j] * vin[j];
#pragma unroll
    for (int off = 16; off > 0; off >>= 1)
        acc += __shfl_down(acc, off, 32);
    if (lane == 0)
        uout[wave] = (1.0f / (float)NS) / (acc + STABF);
}

// partial column sums over a row chunk (coalesced: lane = column)
__global__ void col_matvec_partial(const float* __restrict__ Kmat, const float* __restrict__ uin,
                                   float* __restrict__ part) {
    int j  = blockIdx.x * blockDim.x + threadIdx.x;
    int c  = blockIdx.y;
    int r0 = c * ROWS_PER_CHUNK;
    const float* p = Kmat + (size_t)r0 * NT + j;
    float acc = 0.f;
#pragma unroll 4
    for (int i = 0; i < ROWS_PER_CHUNK; ++i)
        acc += p[(size_t)i * NT] * uin[r0 + i];
    part[(size_t)c * NT + j] = acc;
}

__global__ void finalize_v(const float* __restrict__ part, float* __restrict__ vout) {
    int j = blockIdx.x * blockDim.x + threadIdx.x;
    float acc = 0.f;
#pragma unroll
    for (int c = 0; c < ROW_CHUNKS; ++c)
        acc += part[(size_t)c * NT + j];
    vout[j] = (1.0f / (float)NT) / (acc + STABF);
}

// ---------- phase 3: coupling, cost recovery, mean reduction ----------
// coupling[i,j] = u[i]*K[i,j]*v[j]; cost = -eps*log(K); accumulates coupling*cost.
// Kin may alias Cout (in-place overwrite of the K scratch with coupling).
__global__ void final_coupling(const float* __restrict__ Kin, const float* __restrict__ u,
                               const float* __restrict__ v, float* __restrict__ Cout,
                               float* __restrict__ partials) {
    __shared__ float red[256];
    size_t base = (size_t)blockIdx.x * (blockDim.x * FIN_ELEMS) + threadIdx.x;
    float acc = 0.f;
#pragma unroll
    for (int e = 0; e < FIN_ELEMS; ++e) {
        size_t idx = base + (size_t)e * blockDim.x;
        int i = (int)(idx >> 13);            // / NT
        int j = (int)(idx & (NT - 1));
        float k    = Kin[idx];
        float cpl  = u[i] * k * v[j];
        float cost = -EPS * __logf(k);       // exact inverse of exp(-cost/eps)
        Cout[idx]  = cpl;
        acc += cpl * cost;
    }
    red[threadIdx.x] = acc;
    __syncthreads();
#pragma unroll
    for (int s = 128; s > 0; s >>= 1) {
        if (threadIdx.x < (unsigned)s) red[threadIdx.x] += red[threadIdx.x + s];
        __syncthreads();
    }
    if (threadIdx.x == 0) partials[blockIdx.x] = red[0];
}

__global__ void reduce_partials(const float* __restrict__ partials, int n, float* __restrict__ out) {
    __shared__ float red[256];
    float acc = 0.f;
    for (int i = threadIdx.x; i < n; i += blockDim.x) acc += partials[i];
    red[threadIdx.x] = acc;
    __syncthreads();
#pragma unroll
    for (int s = 128; s > 0; s >>= 1) {
        if (threadIdx.x < (unsigned)s) red[threadIdx.x] += red[threadIdx.x + s];
        __syncthreads();
    }
    if (threadIdx.x == 0) out[0] = red[0] * (1.0f / ((float)NS * (float)NT));
}

__global__ void copy_uv(const float* __restrict__ u, const float* __restrict__ v,
                        float* __restrict__ out_u, float* __restrict__ out_v) {
    int i = blockIdx.x * blockDim.x + threadIdx.x;
    if (i < NS) out_u[i] = u[i]; else out_v[i - NS] = v[i - NS];
}

extern "C" void kernel_launch(void* const* d_in, const int* in_sizes, int n_in,
                              void* d_out, int out_size, void* d_ws, size_t ws_size,
                              hipStream_t stream) {
    (void)in_sizes; (void)n_in; (void)out_size;
    const float* S = (const float*)d_in[0];
    const float* T = (const float*)d_in[1];
    float* out = (float*)d_out;

    // workspace layout (floats), small arrays always in ws
    float* ws       = (float*)d_ws;
    float* s2       = ws;                                   // NS
    float* t2       = s2 + NS;                              // NT
    float* u        = t2 + NT;                              // NS
    float* v        = u  + NS;                              // NT
    float* colpart  = v  + NT;                              // ROW_CHUNKS * NT
    float* partials = colpart + (size_t)ROW_CHUNKS * NT;    // FIN_BLOCKS
    float* small_end = partials + FIN_BLOCKS;
    size_t small_floats = (size_t)(small_end - ws);

    // big K matrix: in ws if it fits, else reuse the coupling output region
    float* Cout = out + 1;                                  // coupling region of d_out
    size_t need_with_K = (small_floats + (size_t)NS * NT) * sizeof(float);
    float* Kmat = (ws_size >= need_with_K) ? small_end : Cout;

    // phase 0: norms
    norms_kernel<<<(NS + NT) / 256, 256, 0, stream>>>(S, T, s2, t2);

    // phase 1: Gibbs kernel via WMMA (one 16x16 tile per wave, 8 waves/block)
    {
        int tiles = (NS / 16) * (NT / 16);                  // 262144
        gibbs_wmma_kernel<<<tiles / 8, 256, 0, stream>>>(S, T, s2, t2, Kmat);
    }

    // phase 2: Sinkhorn iterations
    init_uv<<<(NS + NT) / 256, 256, 0, stream>>>(u, v);
    for (int it = 0; it < N_ITER; ++it) {
        row_matvec_u<<<NS / 8, 256, 0, stream>>>(Kmat, v, u);
        dim3 g(NT / 256, ROW_CHUNKS);
        col_matvec_partial<<<g, 256, 0, stream>>>(Kmat, u, colpart);
        finalize_v<<<NT / 256, 256, 0, stream>>>(colpart, v);
    }

    // phase 3: coupling + distance + u/v outputs
    final_coupling<<<(unsigned)FIN_BLOCKS, 256, 0, stream>>>(Kmat, u, v, Cout, partials);
    reduce_partials<<<1, 256, 0, stream>>>(partials, (int)FIN_BLOCKS, out);
    copy_uv<<<(NS + NT) / 256, 256, 0, stream>>>(
        u, v, out + 1 + (size_t)NS * NT, out + 1 + (size_t)NS * NT + NS);
}
